// Attention_80479097193095
// MI455X (gfx1250) — matrix operand order repaired and checked by reading, not compiled, not run on hardware
//
#include <hip/hip_runtime.h>
#include <hip/hip_bf16.h>

typedef __attribute__((ext_vector_type(16))) _Float16 v16h;
typedef __attribute__((ext_vector_type(8)))  _Float16 v8h;
typedef __attribute__((ext_vector_type(8)))  float    v8f;

#define B_    2
#define S_    4096
#define D_    1024
#define KVH   4
#define HD    64
#define KVD   256
#define NROWS (B_*S_)   /* 8192 */
#define QKVC  768       /* qeff(256) | k(256) | v(256) */

union V16U { v16h v; v8h h[2]; };

#define WMMA_F16(A, Bm, C) \
  __builtin_amdgcn_wmma_f32_16x16x32_f16(false, (A), false, (Bm), (short)0, (C), false, false)

// A-operand (16xK f16, K-chunk of 32): lane m=lane&15, kbase=(lane>>4)*8,
// holds k in [kbase..kbase+7] and [kbase+16..kbase+23]  -> two b128 loads.
__device__ __forceinline__ v16h loadA16(const _Float16* base, int ld, int row0, int k0, int lane) {
  const int row = row0 + (lane & 15);
  const int kb  = k0 + ((lane >> 4) << 3);
  const _Float16* p = base + (long)row * ld + kb;
  V16U u;
  u.h[0] = *(const v8h*)(p);
  u.h[1] = *(const v8h*)(p + 16);
  return u.v;
}

// B-operand (32x16 f16) from an N-major ("transposed") matrix baseT[n][k]:
// lane n=lane&15, kbase=(lane>>4)*8: k in [kbase..kbase+7] and [kbase+16..kbase+23] -> two b128 loads.
__device__ __forceinline__ v16h loadB16t(const _Float16* baseT, int ld, int n0, int k0, int lane) {
  const int n  = n0 + (lane & 15);
  const int kb = k0 + ((lane >> 4) << 3);
  const _Float16* p = baseT + (long)n * ld + kb;
  V16U u;
  u.h[0] = *(const v8h*)(p);
  u.h[1] = *(const v8h*)(p + 16);
  return u.v;
}

__global__ void k_cvt_x(const float* __restrict__ x, _Float16* __restrict__ xh, int n) {
  for (int i = blockIdx.x * blockDim.x + threadIdx.x; i < n; i += gridDim.x * blockDim.x)
    xh[i] = (_Float16)x[i];
}

// Wt[768][1024] (N-major): rows 0-255 group-folded wq*(1/64), 256-511 wk, 512-767 wv.
// wot[1024][256] (N-major) from wo[256][1024].
__global__ void k_prep_w(const float* __restrict__ wq, const float* __restrict__ wk,
                         const float* __restrict__ wv, const float* __restrict__ wo,
                         _Float16* __restrict__ wt, _Float16* __restrict__ wot) {
  const int NW = QKVC * D_;
  const int NO = D_ * KVD;
  for (int i = blockIdx.x * blockDim.x + threadIdx.x; i < NW + NO; i += gridDim.x * blockDim.x) {
    if (i < NW) {
      int c = i / D_, k = i % D_;
      float v;
      if (c < 256) {
        int h = c >> 6, j = c & 63;
        float s = 0.f;
#pragma unroll
        for (int g = 0; g < 4; ++g) s += wq[k * D_ + ((g * 4 + h) * 64 + j)];
        v = s * (1.0f / 64.0f);   // fold both 1/sqrt(HD) scales into Q
      } else if (c < 512) {
        v = wk[k * KVD + (c - 256)];
      } else {
        v = wv[k * KVD + (c - 512)];
      }
      wt[(long)c * D_ + k] = (_Float16)v;
    } else {
      int i2 = i - NW;
      int n = i2 >> 8, c = i2 & 255;
      wot[n * KVD + c] = (_Float16)wo[c * D_ + n];
    }
  }
}

// Fused QKV projection (x@Wt) + interleaved RoPE epilogue.
// 32x64 per wave; all 6 operand loads of a K-step issued before the 8 WMMAs.
__global__ void k_qkv(const _Float16* __restrict__ xh, const _Float16* __restrict__ wt,
                      const float* __restrict__ fcos, const float* __restrict__ fsin,
                      _Float16* __restrict__ qb, _Float16* __restrict__ kbuf,
                      _Float16* __restrict__ vt) {
  const int lane = threadIdx.x & 31;
  const int wave = blockIdx.x * (blockDim.x >> 5) + (threadIdx.x >> 5);
  const int CB = QKVC / 64;                      // 12 col blocks
  const int rb = wave / CB, cb = wave % CB;
  if (rb >= NROWS / 32) return;
  const int r0 = rb * 32, c0 = cb * 64;

  v8f acc[2][4];
#pragma unroll
  for (int rt = 0; rt < 2; ++rt)
#pragma unroll
    for (int ct = 0; ct < 4; ++ct) acc[rt][ct] = (v8f){};

#pragma unroll 2
  for (int kk = 0; kk < D_; kk += 32) {
    if (kk + 32 < D_) {
      __builtin_prefetch(xh + (long)(r0 + (lane & 15)) * D_ + kk + 32, 0, 1);
      __builtin_prefetch(wt + (long)(c0 + (lane & 15)) * D_ + kk + 32, 0, 1);
    }
    v16h a0 = loadA16(xh, D_, r0, kk, lane);
    v16h a1 = loadA16(xh, D_, r0 + 16, kk, lane);
    v16h b[4];
#pragma unroll
    for (int ct = 0; ct < 4; ++ct) b[ct] = loadB16t(wt, D_, c0 + 16 * ct, kk, lane);
#pragma unroll
    for (int ct = 0; ct < 4; ++ct) {
      acc[0][ct] = WMMA_F16(a0, b[ct], acc[0][ct]);
      acc[1][ct] = WMMA_F16(a1, b[ct], acc[1][ct]);
    }
  }

  // C-layout: VGPR i holds row (i + 8*(lane>=16)), col = lane&15. RoPE pairs are
  // adjacent lanes (even/odd head dims): exchange partner via shfl_xor 1.
  const int ncol = lane & 15;
#pragma unroll
  for (int rt = 0; rt < 2; ++rt) {
#pragma unroll
    for (int ct = 0; ct < 4; ++ct) {
      const int c   = c0 + 16 * ct + ncol;
      const int p   = (c & 63) >> 1;
      const float sgn = (c & 1) ? 1.0f : -1.0f;
#pragma unroll
      for (int i = 0; i < 8; ++i) {
        float v  = acc[rt][ct][i];
        float pv = __shfl_xor(v, 1, 32);
        int grow = r0 + 16 * rt + i + ((lane >> 4) << 3);
        int b_   = grow >> 12;                 // row / 4096
        int s    = grow & 4095;
        if (c < 512) {                          // RoPE on q and k only
          float co = fcos[s * 32 + p], si = fsin[s * 32 + p];
          v = v * co + sgn * pv * si;
        }
        _Float16 hv = (_Float16)v;
        if (c < 256) {
          int h = c >> 6, j = c & 63;
          qb[(((long)(b_ * 4 + h) * S_ + s) << 6) + j] = hv;
        } else if (c < 512) {
          int cc = c - 256, h = cc >> 6, j = cc & 63;
          kbuf[(((long)(b_ * 4 + h) * S_ + s) << 6) + j] = hv;
        } else {
          int cc = c - 512, h = cc >> 6, j = cc & 63;
          vt[((long)((b_ * 4 + h) * 64 + j)) * S_ + s] = hv;   // V stored d-major
        }
      }
    }
  }
}

// Flash attention: one wave per (b,h, 32-query block). K/V B-operands are all
// loaded before their WMMA groups and reused across both 16-row Q tiles.
__global__ void __launch_bounds__(32) k_attn(const _Float16* __restrict__ qb,
                                             const _Float16* __restrict__ kb,
                                             const _Float16* __restrict__ vt,
                                             _Float16* __restrict__ ob) {
  __shared__ __align__(16) _Float16 pbuf[32 * 32];
  const int lane = threadIdx.x & 31;
  const int qt = blockIdx.x & (S_ / 32 - 1);   // 128 query blocks
  const int bh = blockIdx.x >> 7;
  const int q0 = qt * 32;
  const _Float16* Q  = qb + (long)bh * S_ * 64;
  const _Float16* K  = kb + (long)bh * S_ * 64;
  const _Float16* Vt = vt + (long)bh * 64 * S_;

  // Q block (32x64, pre-scaled by 1/64) resident: 4 A operands.
  v16h qa[2][2];
#pragma unroll
  for (int rt = 0; rt < 2; ++rt) {
    qa[rt][0] = loadA16(Q, 64, q0 + 16 * rt, 0, lane);
    qa[rt][1] = loadA16(Q, 64, q0 + 16 * rt, 32, lane);
  }

  v8f o[2][4];
  float mi[2][8], li[2][8];
#pragma unroll
  for (int rt = 0; rt < 2; ++rt) {
#pragma unroll
    for (int ct = 0; ct < 4; ++ct) o[rt][ct] = (v8f){};
#pragma unroll
    for (int i = 0; i < 8; ++i) { mi[rt][i] = -1e30f; li[rt][i] = 0.f; }
  }

  const int rowHalf = (lane >> 4) << 3;
  const int ncol = lane & 15;

  for (int kbk = 0; kbk <= q0 + 31; kbk += 32) {     // causal upper bound
    // ---- scores: load all 4 K B-operands, then 8 WMMAs ----
    v16h kB[2][2];
#pragma unroll
    for (int ks = 0; ks < 2; ++ks) {
      kB[ks][0] = loadB16t(K, 64, kbk + 16 * ks, 0, lane);
      kB[ks][1] = loadB16t(K, 64, kbk + 16 * ks, 32, lane);
    }
    v8f sc[2][2];
#pragma unroll
    for (int rt = 0; rt < 2; ++rt) { sc[rt][0] = (v8f){}; sc[rt][1] = (v8f){}; }
#pragma unroll
    for (int ks = 0; ks < 2; ++ks) {
#pragma unroll
      for (int rt = 0; rt < 2; ++rt) {
        sc[rt][ks] = WMMA_F16(qa[rt][0], kB[ks][0], sc[rt][ks]);
        sc[rt][ks] = WMMA_F16(qa[rt][1], kB[ks][1], sc[rt][ks]);
      }
    }
    // ---- prefetch V for this block while softmax runs ----
    v16h vB[4];
#pragma unroll
    for (int ct = 0; ct < 4; ++ct) vB[ct] = loadB16t(Vt, S_, 16 * ct, kbk, lane);

    // ---- causal mask + online softmax (per row tile) ----
    const int key0 = kbk + ncol;
    const int key1 = kbk + 16 + ncol;
#pragma unroll
    for (int rt = 0; rt < 2; ++rt) {
#pragma unroll
      for (int i = 0; i < 8; ++i) {
        const int qrow = q0 + 16 * rt + rowHalf + i;
        float v0 = (key0 <= qrow) ? sc[rt][0][i] : -1e30f;
        float v1 = (key1 <= qrow) ? sc[rt][1][i] : -1e30f;
        float mloc = fmaxf(v0, v1);
#pragma unroll
        for (int mm = 1; mm <= 8; mm <<= 1) mloc = fmaxf(mloc, __shfl_xor(mloc, mm, 32));
        float mnew = fmaxf(mi[rt][i], mloc);
        float corr = __expf(mi[rt][i] - mnew);
        float p0 = __expf(v0 - mnew);
        float p1 = __expf(v1 - mnew);
        float ls = p0 + p1;
#pragma unroll
        for (int mm = 1; mm <= 8; mm <<= 1) ls += __shfl_xor(ls, mm, 32);
        li[rt][i] = li[rt][i] * corr + ls;
        mi[rt][i] = mnew;
        o[rt][0][i] *= corr; o[rt][1][i] *= corr; o[rt][2][i] *= corr; o[rt][3][i] *= corr;
        const int prow = 16 * rt + rowHalf + i;
        pbuf[prow * 32 + ncol]      = (_Float16)p0;
        pbuf[prow * 32 + 16 + ncol] = (_Float16)p1;
      }
    }
    __syncthreads();
    // ---- reload P (2 x 16x32) in WMMA A-layout ----
    v16h pa[2];
    {
      const int row = lane & 15;
      const int kbl = (lane >> 4) << 3;
#pragma unroll
      for (int rt = 0; rt < 2; ++rt) {
        V16U u;
        u.h[0] = *(const v8h*)(&pbuf[(16 * rt + row) * 32 + kbl]);
        u.h[1] = *(const v8h*)(&pbuf[(16 * rt + row) * 32 + kbl + 16]);
        pa[rt] = u.v;
      }
    }
    __syncthreads();
    // ---- P @ V: 8 WMMAs on already-resident V operands ----
#pragma unroll
    for (int ct = 0; ct < 4; ++ct) {
      o[0][ct] = WMMA_F16(pa[0], vB[ct], o[0][ct]);
      o[1][ct] = WMMA_F16(pa[1], vB[ct], o[1][ct]);
    }
  }

  const int b_ = bh >> 2, h = bh & 3;
#pragma unroll
  for (int rt = 0; rt < 2; ++rt) {
#pragma unroll
    for (int i = 0; i < 8; ++i) {
      const int s = q0 + 16 * rt + rowHalf + i;
      const float inv = 1.0f / li[rt][i];
      _Float16* orow = ob + ((long)(b_ * S_ + s) * KVD) + h * 64;
      orow[ncol]      = (_Float16)(o[rt][0][i] * inv);
      orow[16 + ncol] = (_Float16)(o[rt][1][i] * inv);
      orow[32 + ncol] = (_Float16)(o[rt][2][i] * inv);
      orow[48 + ncol] = (_Float16)(o[rt][3][i] * inv);
    }
  }
}

// Final projection: attn_out(8192x256) @ wo -> fp32 d_out. 32x64 per wave.
__global__ void k_out(const _Float16* __restrict__ ob, const _Float16* __restrict__ wot,
                      float* __restrict__ out) {
  const int lane = threadIdx.x & 31;
  const int wave = blockIdx.x * (blockDim.x >> 5) + (threadIdx.x >> 5);
  const int CB = D_ / 64;                        // 16 col blocks
  const int rb = wave / CB, cb = wave % CB;
  if (rb >= NROWS / 32) return;
  const int r0 = rb * 32, c0 = cb * 64;

  v8f acc[2][4];
#pragma unroll
  for (int rt = 0; rt < 2; ++rt)
#pragma unroll
    for (int ct = 0; ct < 4; ++ct) acc[rt][ct] = (v8f){};

#pragma unroll
  for (int kk = 0; kk < KVD; kk += 32) {
    if (kk + 32 < KVD) {
      __builtin_prefetch(ob  + (long)(r0 + (lane & 15)) * KVD + kk + 32, 0, 1);
      __builtin_prefetch(wot + (long)(c0 + (lane & 15)) * KVD + kk + 32, 0, 1);
    }
    v16h a0 = loadA16(ob, KVD, r0, kk, lane);
    v16h a1 = loadA16(ob, KVD, r0 + 16, kk, lane);
    v16h b[4];
#pragma unroll
    for (int ct = 0; ct < 4; ++ct) b[ct] = loadB16t(wot, KVD, c0 + 16 * ct, kk, lane);
#pragma unroll
    for (int ct = 0; ct < 4; ++ct) {
      acc[0][ct] = WMMA_F16(a0, b[ct], acc[0][ct]);
      acc[1][ct] = WMMA_F16(a1, b[ct], acc[1][ct]);
    }
  }
  const int ncol = lane & 15;
#pragma unroll
  for (int rt = 0; rt < 2; ++rt) {
#pragma unroll
    for (int ct = 0; ct < 4; ++ct) {
      const int c = c0 + 16 * ct + ncol;
#pragma unroll
      for (int i = 0; i < 8; ++i) {
        int grow = r0 + 16 * rt + i + ((lane >> 4) << 3);
        out[(long)grow * D_ + c] = acc[rt][ct][i];
      }
    }
  }
}

extern "C" void kernel_launch(void* const* d_in, const int* in_sizes, int n_in,
                              void* d_out, int out_size, void* d_ws, size_t ws_size,
                              hipStream_t stream) {
  (void)in_sizes; (void)n_in; (void)out_size; (void)ws_size;
  const float* x    = (const float*)d_in[0];
  const float* wq   = (const float*)d_in[1];
  const float* wk   = (const float*)d_in[2];
  const float* wv   = (const float*)d_in[3];
  const float* wo   = (const float*)d_in[4];
  const float* fcos = (const float*)d_in[5];
  const float* fsin = (const float*)d_in[6];
  float* out = (float*)d_out;

  char* ws = (char*)d_ws;
  size_t off = 0;
  auto carve = [&](size_t bytes) {
    char* p = ws + off;
    off = (off + bytes + 255) & ~(size_t)255;
    return p;
  };
  _Float16* xh  = (_Float16*)carve((size_t)NROWS * D_ * 2);          // 16 MB
  _Float16* wt  = (_Float16*)carve((size_t)QKVC * D_ * 2);           // 1.5 MB
  _Float16* wot = (_Float16*)carve((size_t)D_ * KVD * 2);            // 0.5 MB
  _Float16* qb  = (_Float16*)carve((size_t)B_ * KVH * S_ * HD * 2);  // 4 MB
  _Float16* kb  = (_Float16*)carve((size_t)B_ * KVH * S_ * HD * 2);  // 4 MB
  _Float16* vt  = (_Float16*)carve((size_t)B_ * KVH * S_ * HD * 2);  // 4 MB
  _Float16* ob  = (_Float16*)carve((size_t)NROWS * KVD * 2);         // 4 MB

  k_cvt_x<<<4096, 256, 0, stream>>>(x, xh, NROWS * D_);
  k_prep_w<<<2048, 256, 0, stream>>>(wq, wk, wv, wo, wt, wot);

  int qkv_waves = (NROWS / 32) * (QKVC / 64);    // 3072
  k_qkv<<<qkv_waves / 8, 256, 0, stream>>>(xh, wt, fcos, fsin, qb, kb, vt);

  k_attn<<<B_ * KVH * (S_ / 32), 32, 0, stream>>>(qb, kb, vt, ob);

  int out_waves = (NROWS / 32) * (D_ / 64);      // 4096
  k_out<<<out_waves / 8, 256, 0, stream>>>(ob, wot, out);
}
